// WindowAttention_33380485824546
// MI455X (gfx1250) — matrix operand order repaired and checked by reading, not compiled, not run on hardware
//
#include <hip/hip_runtime.h>
#include <hip/hip_bf16.h>
#include <math.h>

// ---------------------------------------------------------------------------
// Swin window attention for MI455X (gfx1250), wave32 + WMMA f16 (f32 accum).
//   Stage 0: convert qkv_w / proj_w to f16 (tiny, one-time per launch)
//   Stage 1: qkv = x @ qkv_w.T + qkv_b           (WMMA, f16 row-major to ws)
//   Stage 2: per (window, head) attention        (WMMA QK^T, softmax, WMMA PV)
//   Stage 3: out = ctx @ proj_w.T + proj_b       (WMMA, f32 out)
// ---------------------------------------------------------------------------

typedef __attribute__((ext_vector_type(16))) _Float16 v16h;
typedef __attribute__((ext_vector_type(8)))  _Float16 v8h;
typedef __attribute__((ext_vector_type(8)))  float    v8f;

#define DIMC     192
#define NHEADS   6
#define HDIM     32
#define NTOK     49
#define NWIN     64
#define BWIN     4096                 // total windows
#define MROWS    (BWIN * NTOK)        // 200704 rows
#define QKV_COLS (3 * DIMC)           // 576

static __device__ __forceinline__ v8f wmma16(v16h a, v16h b, v8f c) {
  // emits v_wmma_f32_16x16x32_f16
  return __builtin_amdgcn_wmma_f32_16x16x32_f16(false, a, false, b, (short)0, c,
                                                false, false);
}

static __device__ __forceinline__ v16h cat8(v8h lo, v8h hi) {
  return __builtin_shufflevector(lo, hi, 0, 1, 2, 3, 4, 5, 6, 7,
                                 8, 9, 10, 11, 12, 13, 14, 15);
}

// ---------------------------------------------------------------------------
// Stage 0: weight conversion fp32 -> f16 (both weight matrices).
// ---------------------------------------------------------------------------
__global__ __launch_bounds__(256)
void cvt_weights_kernel(const float* __restrict__ qkv_w,
                        const float* __restrict__ proj_w,
                        _Float16* __restrict__ wq, _Float16* __restrict__ wp) {
  const int i = blockIdx.x * 256 + threadIdx.x;
  if (i < QKV_COLS * DIMC) wq[i] = (_Float16)qkv_w[i];
  if (i < DIMC * DIMC)     wp[i] = (_Float16)proj_w[i];
}

// ---------------------------------------------------------------------------
// Stage 1: QKV projection. Each wave computes a 16x64 output tile.
// A (x) fp32 -> f16 inline; B weights already f16 (direct v16h loads).
// Output f16, row-major [row][576].
// ---------------------------------------------------------------------------
__global__ __launch_bounds__(256)
void qkv_gemm_kernel(const float* __restrict__ x, const _Float16* __restrict__ wh,
                     const float* __restrict__ bias, _Float16* __restrict__ qkv) {
  const int lane  = threadIdx.x & 31;
  const int wv    = threadIdx.x >> 5;
  const int mtile = blockIdx.x * 8 + wv;        // 16-row tile id
  const int nbase = blockIdx.y * 64;            // 64-col block
  const int nl    = lane & 15;
  const int mrow  = mtile * 16 + nl;            // A row owned by this lane
  const int hb    = (lane < 16) ? 0 : 16;       // A K-half base
  const int kb    = (lane < 16) ? 0 : 16;       // B K-half base

  v8f acc[4] = {};
  for (int kt = 0; kt < 6; ++kt) {
    const int k0 = kt * 32;
    const float* ar = x + (size_t)mrow * DIMC + k0;
    // raw A loads (4 x b128), independent of B loads
    const float4 a00 = *(const float4*)(ar + hb);
    const float4 a01 = *(const float4*)(ar + hb + 4);
    const float4 a10 = *(const float4*)(ar + 8 + hb);
    const float4 a11 = *(const float4*)(ar + 8 + hb + 4);
    // B fragments: contiguous 16 halfs per lane, straight to WMMA
    v16h bf[4];
#pragma unroll
    for (int j = 0; j < 4; ++j)
      bf[j] = *(const v16h*)(wh + (size_t)(nbase + j * 16 + nl) * DIMC + k0 + kb);
    v16h a;
    a[0]  = (_Float16)a00.x; a[1]  = (_Float16)a00.y;
    a[2]  = (_Float16)a00.z; a[3]  = (_Float16)a00.w;
    a[4]  = (_Float16)a01.x; a[5]  = (_Float16)a01.y;
    a[6]  = (_Float16)a01.z; a[7]  = (_Float16)a01.w;
    a[8]  = (_Float16)a10.x; a[9]  = (_Float16)a10.y;
    a[10] = (_Float16)a10.z; a[11] = (_Float16)a10.w;
    a[12] = (_Float16)a11.x; a[13] = (_Float16)a11.y;
    a[14] = (_Float16)a11.z; a[15] = (_Float16)a11.w;
#pragma unroll
    for (int j = 0; j < 4; ++j) acc[j] = wmma16(a, bf[j], acc[j]);
  }

#pragma unroll
  for (int j = 0; j < 4; ++j) {
    const int col = nbase + j * 16 + nl;
    const float bv = bias[col];
#pragma unroll
    for (int i = 0; i < 8; ++i) {
      const int row = mtile * 16 + ((lane < 16) ? i : (8 + i));
      qkv[(size_t)row * QKV_COLS + col] = (_Float16)(acc[j][i] + bv);
    }
  }
}

// ---------------------------------------------------------------------------
// Stage 2: attention for one (window, head) per 128-thread block (4 waves).
// q/k zero-padded [64][32] in LDS; V stored transposed [32][64] so PV
// B-fragments are contiguous v16h LDS loads.
// ---------------------------------------------------------------------------
__global__ __launch_bounds__(128)
void attn_kernel(const _Float16* __restrict__ qkv, const float* __restrict__ mask,
                 const float* __restrict__ rpb, const int* __restrict__ rel,
                 float* __restrict__ attn_out, _Float16* __restrict__ ctx) {
  __shared__ _Float16 qs[64 * 32];
  __shared__ _Float16 ks[64 * 32];
  __shared__ _Float16 vt[32 * 64];   // transposed V: vt[c][n]
  __shared__ _Float16 ps[64 * 64];

  const int bh = blockIdx.x;
  const int b  = bh / NHEADS;
  const int h  = bh - b * NHEADS;
  const int tid = threadIdx.x;

  // q at col h*32, k at 192 + h*32, v at 384 + h*32 within each row of 576
  const _Float16* qb = qkv + (size_t)b * NTOK * QKV_COLS + h * HDIM;
  for (int idx = tid; idx < 64 * 32; idx += 128) {
    const int n = idx >> 5;
    const int c = idx & 31;
    const bool vld = n < NTOK;
    const size_t ro = (size_t)n * QKV_COLS + c;
    qs[idx]         = vld ? qb[ro]            : (_Float16)0.f;
    ks[idx]         = vld ? qb[DIMC + ro]     : (_Float16)0.f;
    vt[c * 64 + n]  = vld ? qb[2 * DIMC + ro] : (_Float16)0.f;
  }
  __syncthreads();

  const int wv   = tid >> 5;
  const int lane = tid & 31;
  const int nl   = lane & 15;
  const int hb   = (lane < 16) ? 0 : 16;
  const int kb   = (lane < 16) ? 0 : 16;

  // ---- S = q @ k^T (K = 32 exactly: one WMMA per 16x16 tile) ----
  v16h aq;
  {
    const _Float16* qr = qs + (wv * 16 + nl) * HDIM;
    aq = cat8(*(const v8h*)(qr + hb), *(const v8h*)(qr + 8 + hb));
  }
  v8f s[4];
#pragma unroll
  for (int j = 0; j < 4; ++j) {
    const v16h bf = *(const v16h*)(ks + (j * 16 + nl) * HDIM + kb);
    v8f z = {};
    s[j] = wmma16(aq, bf, z);
  }

  // ---- scale + bias + mask + row softmax ----
  const float scale = 0.17677669529663687f;   // 1/sqrt(32)
  const int   mi    = b & (NWIN - 1);         // window index within image
#pragma unroll
  for (int i = 0; i < 8; ++i) {
    const int r = wv * 16 + ((lane < 16) ? i : (8 + i));
    float lg[4];
#pragma unroll
    for (int j = 0; j < 4; ++j) {
      const int col = j * 16 + nl;
      if (r < NTOK && col < NTOK) {
        const float bv = rpb[rel[r * NTOK + col] * NHEADS + h];
        const float mk = mask[((size_t)mi * NTOK + r) * NTOK + col];
        lg[j] = s[j][i] * scale + bv + mk;
      } else {
        lg[j] = -INFINITY;
      }
    }
    float mx = fmaxf(fmaxf(lg[0], lg[1]), fmaxf(lg[2], lg[3]));
#pragma unroll
    for (int off = 8; off >= 1; off >>= 1) mx = fmaxf(mx, __shfl_xor(mx, off, 32));
    float p[4];
    float sum = 0.f;
#pragma unroll
    for (int j = 0; j < 4; ++j) {
      p[j] = (lg[j] == -INFINITY) ? 0.f : __expf(lg[j] - mx);
      sum += p[j];
    }
#pragma unroll
    for (int off = 8; off >= 1; off >>= 1) sum += __shfl_xor(sum, off, 32);
    const float inv = (sum > 0.f) ? (1.f / sum) : 0.f;
#pragma unroll
    for (int j = 0; j < 4; ++j) {
      const float pj  = p[j] * inv;
      const int   col = j * 16 + nl;
      if (r < NTOK && col < NTOK)
        attn_out[(((size_t)b * NHEADS + h) * NTOK + r) * NTOK + col] = pj;
      ps[r * 64 + col] = (_Float16)pj;      // zero in padded region
    }
  }
  __syncthreads();

  // ---- O = P @ V (K = 64, zero padded) ----
  v8f o[2] = {};
#pragma unroll
  for (int kt = 0; kt < 2; ++kt) {
    const int k0 = kt * 32;
    const _Float16* pr = ps + (wv * 16 + nl) * 64 + k0;
    const v16h ap = cat8(*(const v8h*)(pr + hb), *(const v8h*)(pr + 8 + hb));
#pragma unroll
    for (int j = 0; j < 2; ++j) {
      const v16h bf = *(const v16h*)(vt + (j * 16 + nl) * 64 + k0 + kb);
      o[j] = wmma16(ap, bf, o[j]);
    }
  }
#pragma unroll
  for (int j = 0; j < 2; ++j) {
#pragma unroll
    for (int i = 0; i < 8; ++i) {
      const int r = wv * 16 + ((lane < 16) ? i : (8 + i));
      if (r < NTOK) {
        const int c = j * 16 + nl;
        ctx[((size_t)b * NTOK + r) * DIMC + h * HDIM + c] = (_Float16)o[j][i];
      }
    }
  }
}

// ---------------------------------------------------------------------------
// Stage 3: output projection. ctx and weights are f16 (direct fragment
// loads, no conversions in the K-loop); fp32 output.
// ---------------------------------------------------------------------------
__global__ __launch_bounds__(256)
void proj_gemm_kernel(const _Float16* __restrict__ ctx, const _Float16* __restrict__ wh,
                      const float* __restrict__ bias, float* __restrict__ out) {
  const int lane  = threadIdx.x & 31;
  const int wv    = threadIdx.x >> 5;
  const int mtile = blockIdx.x * 8 + wv;
  const int nbase = blockIdx.y * 64;
  const int nl    = lane & 15;
  const int mrow  = mtile * 16 + nl;
  const int hb    = (lane < 16) ? 0 : 16;
  const int kb    = (lane < 16) ? 0 : 16;

  v8f acc[4] = {};
  for (int kt = 0; kt < 6; ++kt) {
    const int k0 = kt * 32;
    const _Float16* ar = ctx + (size_t)mrow * DIMC + k0;
    const v16h a = cat8(*(const v8h*)(ar + hb), *(const v8h*)(ar + 8 + hb));
    v16h bf[4];
#pragma unroll
    for (int j = 0; j < 4; ++j)
      bf[j] = *(const v16h*)(wh + (size_t)(nbase + j * 16 + nl) * DIMC + k0 + kb);
#pragma unroll
    for (int j = 0; j < 4; ++j) acc[j] = wmma16(a, bf[j], acc[j]);
  }
#pragma unroll
  for (int j = 0; j < 4; ++j) {
    const int col = nbase + j * 16 + nl;
    const float bv = bias[col];
#pragma unroll
    for (int i = 0; i < 8; ++i) {
      const int row = mtile * 16 + ((lane < 16) ? i : (8 + i));
      out[(size_t)row * DIMC + col] = acc[j][i] + bv;
    }
  }
}

// ---------------------------------------------------------------------------
extern "C" void kernel_launch(void* const* d_in, const int* in_sizes, int n_in,
                              void* d_out, int out_size, void* d_ws, size_t ws_size,
                              hipStream_t stream) {
  const float* x      = (const float*)d_in[0];
  const float* mask   = (const float*)d_in[1];
  const float* qkv_w  = (const float*)d_in[2];
  const float* qkv_b  = (const float*)d_in[3];
  const float* proj_w = (const float*)d_in[4];
  const float* proj_b = (const float*)d_in[5];
  const float* rpb    = (const float*)d_in[6];
  const int*   rel    = (const int*)d_in[7];

  float* out  = (float*)d_out;
  float* attn = out + (size_t)MROWS * DIMC;          // second tuple output

  _Float16* wq    = (_Float16*)d_ws;                       // [576][192] f16
  _Float16* wp    = wq + (size_t)QKV_COLS * DIMC;          // [192][192] f16
  _Float16* qkvws = wp + (size_t)DIMC * DIMC;              // [M][576]  f16
  _Float16* ctxws = qkvws + (size_t)MROWS * QKV_COLS;      // [M][192]  f16

  // Stage 0: weight conversion (147456 elements)
  cvt_weights_kernel<<<dim3((QKV_COLS * DIMC + 255) / 256), 256, 0, stream>>>(
      qkv_w, proj_w, wq, wp);

  // Stage 1: 12544 m-tiles / 8 waves per block = 1568 blocks, 9 n-blocks
  qkv_gemm_kernel<<<dim3(MROWS / 16 / 8, QKV_COLS / 64), 256, 0, stream>>>(
      x, wq, qkv_b, qkvws);

  // Stage 2: one block per (window, head)
  attn_kernel<<<dim3(BWIN * NHEADS), 128, 0, stream>>>(
      qkvws, mask, rpb, rel, attn, ctxws);

  // Stage 3: 1568 blocks, 3 n-blocks
  proj_gemm_kernel<<<dim3(MROWS / 16 / 8, DIMC / 64), 256, 0, stream>>>(
      ctxws, wp, proj_b, out);
}
